// MultiHeadAttention_70755291234837
// MI455X (gfx1250) — matrix operand order repaired and checked by reading, not compiled, not run on hardware
//
#include <hip/hip_runtime.h>

// ---------------------------------------------------------------------------
// CDNA5 (gfx1250, wave32) bf16-WMMA multi-head attention forward.
// Pipeline: QKV GEMM (fp32->bf16, scatter per-head) -> flash attention
// (online softmax, WMMA QK^T and PV) -> output projection GEMM (fp32 out).
// ---------------------------------------------------------------------------

typedef __bf16 v16bf __attribute__((ext_vector_type(16)));
typedef __bf16 v8bf  __attribute__((ext_vector_type(8)));
typedef float  v8f   __attribute__((ext_vector_type(8)));

#define WMMA_BF16(A_, B_, C_) \
  __builtin_amdgcn_wmma_f32_16x16x32_bf16(false, (A_), false, (B_), (short)0, (C_), false, false)

namespace {
constexpr int kD  = 1024;   // d_model
constexpr int kS  = 2048;   // sequence length
constexpr int kH  = 16;     // heads
constexpr int kHD = 64;     // head dim
constexpr int LDK = 40;     // LDS row stride (elems) for 32-wide K tiles (80B, 16B-aligned)
constexpr int LDP = 72;     // LDS row stride (elems) for 64-wide P tiles (144B, 16B-aligned)
}

// A-fragment (16x32 bf16, MxK): lane half 0 holds K{0..7,16..23}, half 1 K{8..15,24..31}.
// p must point at (row base + half*8).
__device__ inline v16bf make_afrag(const __bf16* p) {
  v8bf lo = *(const v8bf*)p;
  v8bf hi = *(const v8bf*)(p + 16);
  v16bf f;
#pragma unroll
  for (int e = 0; e < 8; ++e) { f[e] = lo[e]; f[e + 8] = hi[e]; }
  return f;
}

// B-fragment (32x16 bf16, KxN) from an [N][K] row-major source: lane half 0 holds
// K{0..7,16..23}, half 1 K{8..15,24..31}. p must point at (col base + half*8).
__device__ inline v16bf make_bfrag(const __bf16* p) {
  v8bf lo = *(const v8bf*)p;
  v8bf hi = *(const v8bf*)(p + 16);
  v16bf f;
#pragma unroll
  for (int e = 0; e < 8; ++e) { f[e] = lo[e]; f[e + 8] = hi[e]; }
  return f;
}

// ---------------------------------------------------------------------------
// Kernel 1: qkv = x @ w_qkv^T + b_qkv, scattered into Q[b,h,s,64] (pre-scaled
// by 1/sqrt(64)), K[b,h,s,64], V^T[b,h,64,s], all bf16.
// ---------------------------------------------------------------------------
__global__ __launch_bounds__(256) void mha_qkv_gemm(
    const float* __restrict__ x, const float* __restrict__ wqkv,
    const float* __restrict__ bqkv,
    __bf16* __restrict__ Qb, __bf16* __restrict__ Kb, __bf16* __restrict__ Vt) {
  __shared__ __bf16 sA[128 * LDK];
  __shared__ __bf16 sB[128 * LDK];
  const int tid = threadIdx.x;
  const int lane = tid & 31;
  const int wave = tid >> 5;
  const int wm = wave & 1;        // 2 wave-rows of 64
  const int wn = wave >> 1;       // 4 wave-cols of 32
  const int half = lane >> 4;
  const int c = lane & 15;
  const int m0 = blockIdx.x * 128;
  const int n0 = blockIdx.y * 128;

  const v8f vzero = {0.f, 0.f, 0.f, 0.f, 0.f, 0.f, 0.f, 0.f};
  v8f acc[4][2];
#pragma unroll
  for (int i = 0; i < 4; ++i)
#pragma unroll
    for (int j = 0; j < 2; ++j) acc[i][j] = vzero;

  const int lr = tid >> 3;        // 0..31
  const int lc = (tid & 7) * 4;   // 0..28

  for (int k0 = 0; k0 < kD; k0 += 32) {
    __syncthreads();
#pragma unroll
    for (int it = 0; it < 4; ++it) {
      const int r = lr + it * 32;
      float4 av = *(const float4*)(x + (size_t)(m0 + r) * kD + k0 + lc);
      float4 wv = *(const float4*)(wqkv + (size_t)(n0 + r) * kD + k0 + lc);
      __bf16* pa = &sA[r * LDK + lc];
      pa[0] = (__bf16)av.x; pa[1] = (__bf16)av.y; pa[2] = (__bf16)av.z; pa[3] = (__bf16)av.w;
      __bf16* pb = &sB[r * LDK + lc];
      pb[0] = (__bf16)wv.x; pb[1] = (__bf16)wv.y; pb[2] = (__bf16)wv.z; pb[3] = (__bf16)wv.w;
    }
    __syncthreads();
    v16bf afr[4], bfr[2];
#pragma unroll
    for (int i = 0; i < 4; ++i)
      afr[i] = make_afrag(&sA[(wm * 64 + i * 16 + c) * LDK + half * 8]);
#pragma unroll
    for (int j = 0; j < 2; ++j)
      bfr[j] = make_bfrag(&sB[(wn * 32 + j * 16 + c) * LDK + half * 8]);
#pragma unroll
    for (int i = 0; i < 4; ++i)
#pragma unroll
      for (int j = 0; j < 2; ++j) acc[i][j] = WMMA_BF16(afr[i], bfr[j], acc[i][j]);
  }

  // Epilogue: bias + scatter to per-head layouts.
#pragma unroll
  for (int j = 0; j < 2; ++j) {
    const int n = n0 + wn * 32 + j * 16 + c;      // 0..3071
    const float bias = bqkv[n];
    const int h = n / 192;
    const int within = n % 192;
    const int part = within / 64;                 // 0=Q 1=K 2=V
    const int d = within % 64;
#pragma unroll
    for (int i = 0; i < 4; ++i) {
#pragma unroll
      for (int v = 0; v < 8; ++v) {
        const int m = m0 + wm * 64 + i * 16 + v + half * 8;
        const int b = m >> 11;                    // m / 2048
        const int s = m & (kS - 1);
        const float val = acc[i][j][v] + bias;
        const size_t qk = (((size_t)(b * kH + h) * kS) + s) * kHD + d;
        if (part == 0)
          Qb[qk] = (__bf16)(val * 0.125f);        // fold 1/sqrt(64) into Q
        else if (part == 1)
          Kb[qk] = (__bf16)val;
        else
          Vt[(((size_t)(b * kH + h) * kHD) + d) * kS + s] = (__bf16)val;
      }
    }
  }
}

// ---------------------------------------------------------------------------
// Kernel 2: flash attention. grid = (S/128, B*H); 8 waves/WG, 16 q-rows/wave.
// ---------------------------------------------------------------------------
__global__ __launch_bounds__(256) void mha_flash_attn(
    const __bf16* __restrict__ Qb, const __bf16* __restrict__ Kb,
    const __bf16* __restrict__ Vt, __bf16* __restrict__ vals) {
  __shared__ __bf16 sP[8 * 16 * LDP];             // per-wave private P tiles
  const int tid = threadIdx.x;
  const int lane = tid & 31;
  const int wave = tid >> 5;
  const int half = lane >> 4;
  const int c = lane & 15;
  const int bh = blockIdx.y;                      // b*16 + h
  const int q0 = blockIdx.x * 128 + wave * 16;
  const __bf16* Qh = Qb + (size_t)bh * kS * kHD;
  const __bf16* Kh = Kb + (size_t)bh * kS * kHD;
  const __bf16* Vh = Vt + (size_t)bh * kHD * kS;
  __bf16* myP = &sP[wave * 16 * LDP];

  // Q fragments live in registers for the whole kernel (d = 2 k-steps of 32).
  v16bf qf[2];
#pragma unroll
  for (int ks = 0; ks < 2; ++ks)
    qf[ks] = make_afrag(Qh + (size_t)(q0 + c) * kHD + ks * 32 + half * 8);

  const v8f vzero = {0.f, 0.f, 0.f, 0.f, 0.f, 0.f, 0.f, 0.f};
  v8f o[4];
#pragma unroll
  for (int t = 0; t < 4; ++t) o[t] = vzero;
  float mrow[8], lrow[8];                          // stats for row (v + 8*half)
#pragma unroll
  for (int v = 0; v < 8; ++v) { mrow[v] = -3.0e38f; lrow[v] = 0.f; }

  for (int j = 0; j < kS; j += 64) {
    // scores S[16 x 64] = Q * K^T (pre-scaled), 4 N-tiles x 2 k-steps.
    v8f sc[4];
#pragma unroll
    for (int t = 0; t < 4; ++t) {
      v8f a = vzero;
#pragma unroll
      for (int ks = 0; ks < 2; ++ks) {
        v16bf kf = make_bfrag(Kh + (size_t)(j + t * 16 + c) * kHD + ks * 32 + half * 8);
        a = WMMA_BF16(qf[ks], kf, a);
      }
      sc[t] = a;
    }
    // online softmax: row max/sum reduced across the 16 lanes owning each row.
    float rm[8], alpha[8], rs[8];
#pragma unroll
    for (int v = 0; v < 8; ++v)
      rm[v] = fmaxf(fmaxf(sc[0][v], sc[1][v]), fmaxf(sc[2][v], sc[3][v]));
    for (int off = 1; off < 16; off <<= 1)
#pragma unroll
      for (int v = 0; v < 8; ++v)
        rm[v] = fmaxf(rm[v], __shfl_xor(rm[v], off, 32));
#pragma unroll
    for (int v = 0; v < 8; ++v) {
      const float mn = fmaxf(mrow[v], rm[v]);
      alpha[v] = __expf(mrow[v] - mn);
      mrow[v] = mn;
      rs[v] = 0.f;
    }
#pragma unroll
    for (int t = 0; t < 4; ++t) {
#pragma unroll
      for (int v = 0; v < 8; ++v) {
        const float p = __expf(sc[t][v] - mrow[v]);
        rs[v] += p;
        myP[(v + half * 8) * LDP + t * 16 + c] = (__bf16)p;  // C-layout -> LDS
        o[t][v] *= alpha[v];
      }
    }
    for (int off = 1; off < 16; off <<= 1)
#pragma unroll
      for (int v = 0; v < 8; ++v) rs[v] += __shfl_xor(rs[v], off, 32);
#pragma unroll
    for (int v = 0; v < 8; ++v) lrow[v] = lrow[v] * alpha[v] + rs[v];

    // wave-private tile: only need own DS stores complete, no barrier.
    asm volatile("s_wait_dscnt 0" ::: "memory");

    // O[16 x 64] += P[16 x 64] * V[64 keys x 64 d] (B from V^T, row-major in K).
#pragma unroll
    for (int ks = 0; ks < 2; ++ks) {
      v16bf pf = make_afrag(&myP[c * LDP + ks * 32 + half * 8]);
#pragma unroll
      for (int t = 0; t < 4; ++t) {
        v16bf vf = make_bfrag(Vh + (size_t)(t * 16 + c) * kS + j + ks * 32 + half * 8);
        o[t] = WMMA_BF16(pf, vf, o[t]);
      }
    }
  }

  // normalize and store into [B,S,d_model] (head-major columns) as bf16.
  const int b = bh >> 4;
  const int h = bh & 15;
#pragma unroll
  for (int t = 0; t < 4; ++t) {
#pragma unroll
    for (int v = 0; v < 8; ++v) {
      const int row = q0 + v + half * 8;
      vals[((size_t)(b * kS + row)) * kD + h * 64 + t * 16 + c] =
          (__bf16)(o[t][v] / lrow[v]);
    }
  }
}

// ---------------------------------------------------------------------------
// Kernel 3: out = vals @ w_fc^T + b_fc (fp32 output).
// ---------------------------------------------------------------------------
__global__ __launch_bounds__(256) void mha_fc_gemm(
    const __bf16* __restrict__ vals, const float* __restrict__ wfc,
    const float* __restrict__ bfc, float* __restrict__ out) {
  __shared__ __bf16 sA[128 * LDK];
  __shared__ __bf16 sB[128 * LDK];
  const int tid = threadIdx.x;
  const int lane = tid & 31;
  const int wave = tid >> 5;
  const int wm = wave & 1;
  const int wn = wave >> 1;
  const int half = lane >> 4;
  const int c = lane & 15;
  const int m0 = blockIdx.x * 128;
  const int n0 = blockIdx.y * 128;

  const v8f vzero = {0.f, 0.f, 0.f, 0.f, 0.f, 0.f, 0.f, 0.f};
  v8f acc[4][2];
#pragma unroll
  for (int i = 0; i < 4; ++i)
#pragma unroll
    for (int j = 0; j < 2; ++j) acc[i][j] = vzero;

  const int ar = tid >> 2;        // 0..63
  const int ac = (tid & 3) * 8;   // 0,8,16,24 (bf16 b128 loads)
  const int br = tid >> 3;        // 0..31
  const int bc = (tid & 7) * 4;   // fp32 float4 loads

  for (int k0 = 0; k0 < kD; k0 += 32) {
    __syncthreads();
#pragma unroll
    for (int it = 0; it < 2; ++it) {
      const int r = ar + it * 64;
      v8bf av = *(const v8bf*)(vals + (size_t)(m0 + r) * kD + k0 + ac);
      *(v8bf*)&sA[r * LDK + ac] = av;
    }
#pragma unroll
    for (int it = 0; it < 4; ++it) {
      const int r = br + it * 32;
      float4 wv = *(const float4*)(wfc + (size_t)(n0 + r) * kD + k0 + bc);
      __bf16* pb = &sB[r * LDK + bc];
      pb[0] = (__bf16)wv.x; pb[1] = (__bf16)wv.y; pb[2] = (__bf16)wv.z; pb[3] = (__bf16)wv.w;
    }
    __syncthreads();
    v16bf afr[4], bfr[2];
#pragma unroll
    for (int i = 0; i < 4; ++i)
      afr[i] = make_afrag(&sA[(wm * 64 + i * 16 + c) * LDK + half * 8]);
#pragma unroll
    for (int j = 0; j < 2; ++j)
      bfr[j] = make_bfrag(&sB[(wn * 32 + j * 16 + c) * LDK + half * 8]);
#pragma unroll
    for (int i = 0; i < 4; ++i)
#pragma unroll
      for (int j = 0; j < 2; ++j) acc[i][j] = WMMA_BF16(afr[i], bfr[j], acc[i][j]);
  }

#pragma unroll
  for (int j = 0; j < 2; ++j) {
    const int n = n0 + wn * 32 + j * 16 + c;
    const float bias = bfc[n];
#pragma unroll
    for (int i = 0; i < 4; ++i) {
#pragma unroll
      for (int v = 0; v < 8; ++v) {
        const int m = m0 + wm * 64 + i * 16 + v + half * 8;
        out[(size_t)m * kD + n] = acc[i][j][v] + bias;
      }
    }
  }
}

// ---------------------------------------------------------------------------
// Host launcher. Workspace layout (bytes):
//   [0,8M)   Q  bf16 [B,H,S,64]   (pre-scaled by 1/8)
//   [8M,16M) K  bf16 [B,H,S,64]
//   [16M,24M)V^T bf16 [B,H,64,S]
//   [24M,32M)vals bf16 [B,S,1024]
// ---------------------------------------------------------------------------
extern "C" void kernel_launch(void* const* d_in, const int* in_sizes, int n_in,
                              void* d_out, int out_size, void* d_ws, size_t ws_size,
                              hipStream_t stream) {
  (void)in_sizes; (void)n_in; (void)out_size; (void)ws_size;
  const float* x    = (const float*)d_in[0];
  const float* wqkv = (const float*)d_in[1];
  const float* bqkv = (const float*)d_in[2];
  const float* wfc  = (const float*)d_in[3];
  const float* bfc  = (const float*)d_in[4];
  float* out = (float*)d_out;

  char* ws = (char*)d_ws;
  const size_t perBuf = (size_t)2 * 16 * 2048 * 64;   // elements per Q/K/V buffer
  __bf16* Qb   = (__bf16*)(ws);
  __bf16* Kb   = (__bf16*)(ws + perBuf * 2);
  __bf16* Vt   = (__bf16*)(ws + perBuf * 4);
  __bf16* vals = (__bf16*)(ws + perBuf * 6);

  mha_qkv_gemm<<<dim3(32, 24), 256, 0, stream>>>(x, wqkv, bqkv, Qb, Kb, Vt);
  mha_flash_attn<<<dim3(16, 32), 256, 0, stream>>>(Qb, Kb, Vt, vals);
  mha_fc_gemm<<<dim3(32, 8), 256, 0, stream>>>(vals, wfc, bfc, out);
}
